// DynamicFiltering_41394894799612
// MI455X (gfx1250) — matrix operand order repaired and checked by reading, not compiled, not run on hardware
//
#include <hip/hip_runtime.h>

typedef __attribute__((ext_vector_type(16))) _Float16 v16h;
typedef __attribute__((ext_vector_type(8)))  _Float16 v8h;
typedef __attribute__((ext_vector_type(8)))  float    v8f;
typedef __attribute__((ext_vector_type(4)))  float    v4f;

// Problem constants: b=2, c=64, t=5, h=w=128, K=45
// x layout   : (b, c, t, h, w) f32
// x16 layout : (img=b*t, h, w, cin) f16   -- channels last
// y16 layout : (img, h, w, cout) f16
// ker layout : (b, h, w, t*9) f32
// A-fragment buffers are pre-swizzled to the exact per-lane WMMA layout.

// ---------------- prep: x -> channels-last f16 ----------------
__global__ __launch_bounds__(256)
void prep_x16(const float* __restrict__ x, _Float16* __restrict__ x16)
{
    int gid = blockIdx.x * 256 + threadIdx.x;      // one (img,h,w) per thread, 163840 total
    int img = gid >> 14;
    int hw  = gid & 16383;
    int bb  = img / 5, tt = img % 5;
    const float* xp = x + bb * 655360 + tt * 16384 + hw;   // c stride = 5*128*128 = 81920
    _Float16* op = x16 + gid * 64;
#pragma unroll
    for (int c8 = 0; c8 < 8; ++c8) {
        v8h v;
#pragma unroll
        for (int i = 0; i < 8; ++i)
            v[i] = (_Float16)xp[(c8 * 8 + i) * 81920];
        *(v8h*)(op + c8 * 8) = v;
    }
}

// ---------------- prep: swizzle w1 into A fragments ----------------
// frag index: [mt(4)][kc(18)][lane(32)][e(16)]
// A 16x32 in contiguous K order (B matches): row m = lane&15, K = e + 16*(lane>=16)
__global__ __launch_bounds__(256)
void prep_w1frag(const float* __restrict__ w1, _Float16* __restrict__ wf)
{
    int gid  = blockIdx.x * 256 + threadIdx.x;     // < 36864
    int e    = gid & 15;
    int lane = (gid >> 4) & 31;
    int rest = gid >> 9;                           // mt*18 + kc
    int kc   = rest % 18;
    int mt   = rest / 18;
    int K    = e + ((lane >= 16) ? 16 : 0);
    int tap  = kc >> 1;
    int cin  = ((kc & 1) << 5) + K;
    int r    = tap / 3, s = tap % 3;
    int cout = mt * 16 + (lane & 15);
    wf[gid] = (_Float16)w1[((cout * 64 + cin) * 3 + r) * 3 + s];
}

__global__ __launch_bounds__(256)
void prep_w2frag(const float* __restrict__ w2, _Float16* __restrict__ wf)
{
    int gid  = blockIdx.x * 256 + threadIdx.x;     // < 9216
    int e    = gid & 15;
    int lane = (gid >> 4) & 31;
    int kc   = gid >> 9;                           // 0..17
    int K    = e + ((lane >= 16) ? 16 : 0);
    int tap  = kc >> 1;
    int cin  = ((kc & 1) << 5) + K;
    int r    = tap / 3, s = tap % 3;
    int m    = lane & 15;
    float v  = (m < 9) ? w2[((m * 64 + cin) * 3 + r) * 3 + s] : 0.0f;
    wf[gid] = (_Float16)v;
}

// ---------------- conv1: implicit GEMM, 64cout x 16px per wave ----------------
__global__ __launch_bounds__(128)
void conv1_wmma(const _Float16* __restrict__ x16, const _Float16* __restrict__ wfrag,
                const float* __restrict__ b1, _Float16* __restrict__ y16)
{
    const int lane = threadIdx.x & 31;
    const int tile = blockIdx.x * 4 + (threadIdx.x >> 5);  // 10240 tiles
    const int img  = tile >> 10;                           // 128*8 tiles per image
    const int rem  = tile & 1023;
    const int h    = rem >> 3;
    const int wb   = (rem & 7) << 4;
    const int n    = lane & 15;
    const int hi   = lane >> 4;

    v8f acc[4] = {{}, {}, {}, {}};

#pragma unroll
    for (int kc = 0; kc < 18; ++kc) {
        const int tap = kc >> 1;
        const int dr  = tap / 3 - 1;
        const int ds  = tap % 3 - 1;
        const int cinBase = ((kc & 1) << 5) + (hi << 4);   // B lanes>=16 hold K=16..31
        const int hh = h + dr;
        const int ww = wb + n + ds;

        v16h bfrag = {};                                   // zero padding
        if ((unsigned)hh < 128u && (unsigned)ww < 128u)
            bfrag = *(const v16h*)(x16 + (((img * 128 + hh) * 128 + ww) * 64 + cinBase));

        const _Float16* wp = wfrag + (kc * 32 + lane) * 16;
#pragma unroll
        for (int mt = 0; mt < 4; ++mt) {
            v16h afrag = *(const v16h*)(wp + mt * (18 * 32 * 16));
            acc[mt] = __builtin_amdgcn_wmma_f32_16x16x32_f16(
                false, afrag, false, bfrag, (short)0, acc[mt], false, false);
        }
    }

    _Float16* outp = y16 + ((img * 128 + h) * 128 + wb + n) * 64;
#pragma unroll
    for (int mt = 0; mt < 4; ++mt) {
        const int coutBase = mt * 16 + hi * 8;             // D: lanes>=16 hold M=8..15
        float bias[8];
        *(v4f*)&bias[0] = *(const v4f*)(b1 + coutBase);
        *(v4f*)&bias[4] = *(const v4f*)(b1 + coutBase + 4);
        v8h o;
#pragma unroll
        for (int v = 0; v < 8; ++v) {
            float t = acc[mt][v] + bias[v];
            t = (t >= 0.0f) ? t : 0.2f * t;                // LeakyReLU(0.2)
            o[v] = (_Float16)t;
        }
        *(v8h*)(outp + coutBase) = o;
    }
}

// ---------------- conv2: 9cout x 16px per wave, scatter into ker ----------------
__global__ __launch_bounds__(128)
void conv2_wmma(const _Float16* __restrict__ y16, const _Float16* __restrict__ wfrag,
                const float* __restrict__ b2, float* __restrict__ ker)
{
    const int lane = threadIdx.x & 31;
    const int tile = blockIdx.x * 4 + (threadIdx.x >> 5);
    const int img  = tile >> 10;
    const int rem  = tile & 1023;
    const int h    = rem >> 3;
    const int wb   = (rem & 7) << 4;
    const int n    = lane & 15;
    const int hi   = lane >> 4;

    v8f acc = {};

#pragma unroll
    for (int kc = 0; kc < 18; ++kc) {
        const int tap = kc >> 1;
        const int dr  = tap / 3 - 1;
        const int ds  = tap % 3 - 1;
        const int cinBase = ((kc & 1) << 5) + (hi << 4);
        const int hh = h + dr;
        const int ww = wb + n + ds;

        v16h bfrag = {};
        if ((unsigned)hh < 128u && (unsigned)ww < 128u)
            bfrag = *(const v16h*)(y16 + (((img * 128 + hh) * 128 + ww) * 64 + cinBase));

        v16h afrag = *(const v16h*)(wfrag + (kc * 32 + lane) * 16);
        acc = __builtin_amdgcn_wmma_f32_16x16x32_f16(
            false, afrag, false, bfrag, (short)0, acc, false, false);
    }

    const int bb = img / 5, tt = img % 5;
    const int ww = wb + n;
    float* kp = ker + (((bb * 128 + h) * 128 + ww) * 45 + tt * 9);
    const int m0 = hi * 8;
#pragma unroll
    for (int v = 0; v < 8; ++v) {
        int m = m0 + v;
        if (m < 9) kp[m] = acc[v] + b2[m];
    }
}

// ---------------- dynamic filter: normalize 45 weights + edge-clamped 3x3x5 tap ----------------
__global__ __launch_bounds__(128)
void dynfilt(const float* __restrict__ x, const float* __restrict__ ker,
             float* __restrict__ out)
{
    const int w  = threadIdx.x;
    const int h  = blockIdx.x & 127;
    const int q  = (blockIdx.x >> 7) & 3;      // channel quarter
    const int bb = blockIdx.x >> 9;

    const float* kp = ker + ((bb * 128 + h) * 128 + w) * 45;
    float kf[45];
    float s = 0.0f;
#pragma unroll
    for (int i = 0; i < 45; ++i) { kf[i] = kp[i]; s += kf[i]; }
    const float adj = (1.0f - s) * (1.0f / 45.0f);   // -mean + 1/K
#pragma unroll
    for (int i = 0; i < 45; ++i) kf[i] += adj;

    const int hrow[3] = {(h > 0) ? h - 1 : 0, h, (h < 127) ? h + 1 : 127};
    const int wcol[3] = {(w > 0) ? w - 1 : 0, w, (w < 127) ? w + 1 : 127};

    for (int ci = 0; ci < 16; ++ci) {
        const int c = q * 16 + ci;
        float acc = 0.0f;
        int idx = 0;
#pragma unroll
        for (int t = 0; t < 5; ++t) {
            const float* xb = x + (((bb * 64 + c) * 5 + t) << 14);
#pragma unroll
            for (int di = 0; di < 3; ++di) {
                const float* xr = xb + (hrow[di] << 7);
#pragma unroll
                for (int dj = 0; dj < 3; ++dj)
                    acc += xr[wcol[dj]] * kf[idx++];
            }
        }
        out[((bb * 64 + c) << 14) + (h << 7) + w] = acc;
    }
}

extern "C" void kernel_launch(void* const* d_in, const int* in_sizes, int n_in,
                              void* d_out, int out_size, void* d_ws, size_t ws_size,
                              hipStream_t stream)
{
    (void)in_sizes; (void)n_in; (void)out_size; (void)ws_size;
    const float* x  = (const float*)d_in[0];
    const float* w1 = (const float*)d_in[1];
    const float* b1 = (const float*)d_in[2];
    const float* w2 = (const float*)d_in[3];
    const float* b2 = (const float*)d_in[4];
    float* out = (float*)d_out;

    char* ws = (char*)d_ws;
    _Float16* x16 = (_Float16*)(ws);                 // 20,971,520 B
    _Float16* y16 = (_Float16*)(ws + 20971520);      // 20,971,520 B
    float*    ker = (float*)   (ws + 41943040);      //  5,898,240 B
    _Float16* w1f = (_Float16*)(ws + 47841280);      //     73,728 B
    _Float16* w2f = (_Float16*)(ws + 47915008);      //     18,432 B

    prep_x16   <<<640,  256, 0, stream>>>(x, x16);
    prep_w1frag<<<144,  256, 0, stream>>>(w1, w1f);
    prep_w2frag<<<36,   256, 0, stream>>>(w2, w2f);
    conv1_wmma <<<2560, 128, 0, stream>>>(x16, w1f, b1, y16);
    conv2_wmma <<<2560, 128, 0, stream>>>(y16, w2f, b2, ker);
    dynfilt    <<<1024, 128, 0, stream>>>(x, ker, out);
}
